// GroupQueryAttention_75333726372497
// MI455X (gfx1250) — matrix operand order repaired and checked by reading, not compiled, not run on hardware
//
#include <hip/hip_runtime.h>
#include <stdint.h>

// ---------- CDNA5 WMMA types ----------
typedef __attribute__((ext_vector_type(16))) _Float16 v16h;
typedef __attribute__((ext_vector_type(8)))  float    v8f;
typedef __attribute__((ext_vector_type(4)))  float    f32x4;
typedef __attribute__((ext_vector_type(4)))  _Float16 f16x4;
typedef __attribute__((ext_vector_type(4)))  uint32_t u32x4;

union AFrag { v16h h; uint32_t u[8]; };           // A: per-lane K = 8*half+{0..7}, 16+8*half+{0..7}
union BFrag { v16h h; u32x4 q[2]; };              // B: per-lane col n=lane&15, K = 8*half+{0..7}, 16+8*half+{0..7}

__device__ __forceinline__ v8f wmma_f16(v16h a, v16h b, v8f c) {
  // D = A(16x32 f16) * B(32x16 f16) + C(16x16 f32)
  return __builtin_amdgcn_wmma_f32_16x16x32_f16(false, a, false, b, (short)0, c, false, false);
}

__device__ __forceinline__ void load_afrag(AFrag& a, const _Float16* p) {
#pragma unroll
  for (int v = 0; v < 4; ++v) {
    a.u[v]     = *(const uint32_t*)(p + 2 * v);
    a.u[v + 4] = *(const uint32_t*)(p + 16 + 2 * v);
  }
}
__device__ __forceinline__ void load_bfrag(BFrag& b, const _Float16* p) {
  const u32x4* q = (const u32x4*)p;
  b.q[0] = q[0]; b.q[1] = q[2];
}

// ---------- f32 -> f16 cast (vectorized x4) ----------
__global__ void gqa_cvt4(const float* __restrict__ in, _Float16* __restrict__ out, int n4) {
  int i = blockIdx.x * blockDim.x + threadIdx.x;
  if (i >= n4) return;
  f32x4 v = ((const f32x4*)in)[i];
  f16x4 o;
  o[0] = (_Float16)v[0]; o[1] = (_Float16)v[1];
  o[2] = (_Float16)v[2]; o[3] = (_Float16)v[3];
  ((f16x4*)out)[i] = o;
}

// ---------- WMMA GEMM: out[m][n] = sum_k A[m][k] * W[n][k] ----------
// mode 0: f16 out, row-major [M][N]
// mode 1: f16 out, transposed for V: out[((b*N+col)<<11)+t], b=row>>11, t=row&2047
// mode 2: f32 out + bias, row-major [M][N]
__global__ __launch_bounds__(256) void gqa_gemm(
    const _Float16* __restrict__ A, const _Float16* __restrict__ W,
    const float* __restrict__ bias, _Float16* __restrict__ outh,
    float* __restrict__ outf, int M, int N, int mode) {
  constexpr int K = 1024;
  const int lane = threadIdx.x & 31;
  const int wave = (int)((blockIdx.x * blockDim.x + threadIdx.x) >> 5);
  const int ntiles = N >> 6;                 // 64-wide column tiles per wave
  const int mt = wave / ntiles;
  const int nt = wave - mt * ntiles;
  if (mt * 16 >= M) return;
  const int half = lane >> 4, l16 = lane & 15;
  const int m0 = mt * 16, n0 = nt * 64;

  const _Float16* arow = A + (size_t)(m0 + l16) * K + 8 * half;
  const _Float16* wrow = W + (size_t)(n0 + l16) * K + 8 * half;

  v8f acc[4] = {v8f{}, v8f{}, v8f{}, v8f{}};

  // software pipeline: stage k0's fragments, prefetch k0+32 before the WMMAs
  AFrag a;  BFrag bm[4];
  load_afrag(a, arow);
#pragma unroll
  for (int s = 0; s < 4; ++s) load_bfrag(bm[s], wrow + (size_t)s * (16 * K));

  for (int k0 = 0; k0 < K; k0 += 32) {
    const int kn = (k0 + 32 < K) ? (k0 + 32) : 0;   // clamped: always-valid prefetch
    AFrag an;  BFrag bn[4];
    load_afrag(an, arow + kn);
#pragma unroll
    for (int s = 0; s < 4; ++s) load_bfrag(bn[s], wrow + (size_t)s * (16 * K) + kn);

    // keep the prefetch loads ABOVE the WMMAs: nothing may cross this point
    __builtin_amdgcn_sched_barrier(0);

#pragma unroll
    for (int s = 0; s < 4; ++s) acc[s] = wmma_f16(a.h, bm[s].h, acc[s]);

    a = an;
#pragma unroll
    for (int s = 0; s < 4; ++s) bm[s] = bn[s];
  }

#pragma unroll
  for (int s = 0; s < 4; ++s) {
#pragma unroll
    for (int r = 0; r < 8; ++r) {
      float val = acc[s][r];
      int row = m0 + r + 8 * half;
      int col = n0 + 16 * s + l16;
      if (mode == 2) {
        outf[(size_t)row * N + col] = val + bias[col];
      } else if (mode == 1) {
        int bb = row >> 11, t = row & 2047;
        outh[(((size_t)(bb * N + col)) << 11) + t] = (_Float16)val;
      } else {
        outh[(size_t)row * N + col] = (_Float16)val;
      }
    }
  }
}

// ---------- Flash attention: one wave per (b, head, 16-query tile) ----------
// Q:  [B][T][H][64] f16     Kh: [B][T][G][64] f16
// Vt: [B][G][64][T] f16 (feature-major)   O: [B][T][H*64] f16
__global__ __launch_bounds__(256) void gqa_attn(
    const _Float16* __restrict__ Q, const _Float16* __restrict__ Kh,
    const _Float16* __restrict__ Vt, _Float16* __restrict__ O) {
  __shared__ _Float16 plds[8][16 * 32];      // per-wave P tile (16 q-rows x 32 keys)
  const int lane = threadIdx.x & 31;
  const int w = threadIdx.x >> 5;
  const int wave = blockIdx.x * 8 + w;       // 8192 waves total
  const int qt = wave & 127;
  const int h  = (wave >> 7) & 15;
  const int b  = wave >> 11;
  const int g  = h >> 2;                     // GROUP_SIZE = 4
  const int q0 = qt << 4;
  const int half = lane >> 4, l16 = lane & 15;

  // Q A-fragments for head-dim slices [0,32) and [32,64)
  AFrag qa[2];
  {
    const _Float16* qrow = Q + (((size_t)(b * 2048 + q0 + l16) * 16 + h) << 6) + 8 * half;
    load_afrag(qa[0], qrow);
    load_afrag(qa[1], qrow + 32);
  }

  float mrow[8], lrow[8];
  v8f o[4] = {v8f{}, v8f{}, v8f{}, v8f{}};
#pragma unroll
  for (int r = 0; r < 8; ++r) { mrow[r] = -1e30f; lrow[r] = 0.0f; }

  _Float16* pw = &plds[w][0];
  const _Float16* kb0 = Kh + (((size_t)(b * 2048 + l16) * 4 + g) << 6) + 8 * half;
  const _Float16* vb0 = Vt + (((size_t)(b * 4 + g)) << 17) + ((size_t)l16 << 11) + 8 * half;
  const int kmax = q0 + 16;

  // K fragments: [fs] = 16-key tile 0, [2+fs] = tile 1
  BFrag bk[4];
  {
    const _Float16* kb = kb0;
#pragma unroll
    for (int fs = 0; fs < 2; ++fs) {
      load_bfrag(bk[fs],     kb + 32 * fs);
      load_bfrag(bk[2 + fs], kb + 4096 + 32 * fs);   // +16 keys
    }
  }

  for (int kt0 = 0; kt0 < kmax; kt0 += 32) {
    // ---- S = Q * K^T for two 16-key tiles ----
    v8f s0 = {}, s1 = {};
#pragma unroll
    for (int fs = 0; fs < 2; ++fs) {
      s0 = wmma_f16(qa[fs].h, bk[fs].h, s0);
      s1 = wmma_f16(qa[fs].h, bk[2 + fs].h, s1);
    }

    // ---- prefetch next K tiles + current V tiles (overlap with softmax) ----
    const int ktn = (kt0 + 32 < kmax) ? (kt0 + 32) : 0;
    BFrag bkn[4];
    {
      const _Float16* kb = kb0 + (size_t)ktn * 256;
#pragma unroll
      for (int fs = 0; fs < 2; ++fs) {
        load_bfrag(bkn[fs],     kb + 32 * fs);
        load_bfrag(bkn[2 + fs], kb + 4096 + 32 * fs);
      }
    }
    BFrag bv[4];
    {
      const _Float16* vb = vb0 + kt0;
#pragma unroll
      for (int s = 0; s < 4; ++s) load_bfrag(bv[s], vb + ((size_t)s << 15));  // +16 features
    }

    // prefetch loads must issue before the softmax ALU chain below
    __builtin_amdgcn_sched_barrier(0);

    // ---- online softmax (rows owned per-lane-half; xor<16 shuffles) ----
    const bool need_mask = (kt0 + 31) > q0;
#pragma unroll
    for (int r = 0; r < 8; ++r) {
      float v0 = s0[r] * 0.125f;             // 1/sqrt(64)
      float v1 = s1[r] * 0.125f;
      if (need_mask) {
        int row = q0 + r + 8 * half;
        int key = kt0 + l16;
        if (key > row)      v0 = -1e30f;
        if (key + 16 > row) v1 = -1e30f;
      }
      float mx = fmaxf(v0, v1);
      mx = fmaxf(mx, __shfl_xor(mx, 1, 32));
      mx = fmaxf(mx, __shfl_xor(mx, 2, 32));
      mx = fmaxf(mx, __shfl_xor(mx, 4, 32));
      mx = fmaxf(mx, __shfl_xor(mx, 8, 32));
      float mnew = fmaxf(mrow[r], mx);
      float sc = __expf(mrow[r] - mnew);
      mrow[r] = mnew;
      float p0 = __expf(v0 - mnew);
      float p1 = __expf(v1 - mnew);
      float sum = p0 + p1;
      sum += __shfl_xor(sum, 1, 32);
      sum += __shfl_xor(sum, 2, 32);
      sum += __shfl_xor(sum, 4, 32);
      sum += __shfl_xor(sum, 8, 32);
      lrow[r] = lrow[r] * sc + sum;
      o[0][r] *= sc; o[1][r] *= sc; o[2][r] *= sc; o[3][r] *= sc;
      int m = r + 8 * half;
      pw[m * 32 + l16]      = (_Float16)p0;  // P row-major into LDS
      pw[m * 32 + 16 + l16] = (_Float16)p1;
    }

    asm volatile("s_wait_dscnt 0" ::: "memory");   // LDS RAW before A-fragment reload

    // ---- P reloaded in A layout ----
    AFrag pa;
    load_afrag(pa, pw + l16 * 32 + 8 * half);

    // ---- O += P * V ----
#pragma unroll
    for (int s = 0; s < 4; ++s) o[s] = wmma_f16(pa.h, bv[s].h, o[s]);

#pragma unroll
    for (int i = 0; i < 4; ++i) bk[i] = bkn[i];
  }

  // ---- normalize + store [b][t][h*64+f] ----
#pragma unroll
  for (int r = 0; r < 8; ++r) {
    float inv = 1.0f / lrow[r];
    int row = q0 + r + 8 * half;
    _Float16* orow = O + (((size_t)(b * 2048 + row)) << 10) + h * 64 + l16;
#pragma unroll
    for (int s = 0; s < 4; ++s)
      orow[s * 16] = (_Float16)(o[s][r] * inv);
  }
}

// ---------- host launcher ----------
extern "C" void kernel_launch(void* const* d_in, const int* in_sizes, int n_in,
                              void* d_out, int out_size, void* d_ws, size_t ws_size,
                              hipStream_t stream) {
  const float* x  = (const float*)d_in[0];   // [4][2048][1024]
  const float* Wq = (const float*)d_in[1];   // [16][64][1024]
  const float* Wk = (const float*)d_in[2];   // [4][64][1024]
  const float* Wv = (const float*)d_in[3];   // [4][64][1024]
  const float* Wp = (const float*)d_in[4];   // [1024][1024]
  const float* bp = (const float*)d_in[5];   // [1024]
  float* out = (float*)d_out;

  char* ws = (char*)d_ws;
  _Float16* xh  = (_Float16*)(ws + 0);          // 16 MB
  _Float16* Wqh = (_Float16*)(ws + 16777216);   //  2 MB
  _Float16* Wkh = (_Float16*)(ws + 18874368);   // .5 MB
  _Float16* Wvh = (_Float16*)(ws + 19398656);   // .5 MB
  _Float16* Wph = (_Float16*)(ws + 19922944);   //  2 MB
  _Float16* qh  = (_Float16*)(ws + 22020096);   // 16 MB  [B][T][H][64]
  _Float16* kh  = (_Float16*)(ws + 38797312);   //  4 MB  [B][T][G][64]
  _Float16* vh  = (_Float16*)(ws + 42991616);   //  4 MB  [B][G][64][T]
  _Float16* ah  = (_Float16*)(ws + 47185920);   // 16 MB  [B][T][1024]

  // casts
  gqa_cvt4<<<8192, 256, 0, stream>>>(x,  xh,  2097152);
  gqa_cvt4<<<1024, 256, 0, stream>>>(Wq, Wqh, 262144);
  gqa_cvt4<<<256,  256, 0, stream>>>(Wk, Wkh, 65536);
  gqa_cvt4<<<256,  256, 0, stream>>>(Wv, Wvh, 65536);
  gqa_cvt4<<<1024, 256, 0, stream>>>(Wp, Wph, 262144);

  // projections (M=8192, K=1024); V stored transposed for attention
  gqa_gemm<<<1024, 256, 0, stream>>>(xh, Wqh, nullptr, qh, nullptr, 8192, 1024, 0);
  gqa_gemm<<<256,  256, 0, stream>>>(xh, Wkh, nullptr, kh, nullptr, 8192, 256, 0);
  gqa_gemm<<<256,  256, 0, stream>>>(xh, Wvh, nullptr, vh, nullptr, 8192, 256, 1);

  // attention: 4*16*128 = 8192 waves
  gqa_attn<<<1024, 256, 0, stream>>>(qh, kh, vh, ah);

  // output projection + bias (f32 out)
  gqa_gemm<<<1024, 256, 0, stream>>>(ah, Wph, bp, nullptr, out, 8192, 1024, 2);
}
